// QConv_36627481101119
// MI455X (gfx1250) — compile-verified, matrix operand order checked by reading, not run on hardware
//
#include <hip/hip_runtime.h>

// ---------------------------------------------------------------------------
// Quaternion conv2d as implicit GEMM with CDNA5 bf16 WMMA (gfx1250, wave32).
// M = 256 (Hamilton-expanded out channels), N = B*H*W pixels, K = 256*9.
// Block = 256 threads (8 waves): BLOCK_M=256 x BLOCK_N=64 (one output row).
// Waves arranged 4(M) x 2(N); each wave: 64x32 tile = 4x2 WMMA 16x16 tiles.
// A (weights, bf16 in workspace) staged via async global->LDS (ASYNCcnt).
// ---------------------------------------------------------------------------

typedef __bf16 bf16x16 __attribute__((ext_vector_type(16)));
typedef float  f32x8   __attribute__((ext_vector_type(8)));
typedef int    v4i     __attribute__((vector_size(16)));
typedef v4i __attribute__((address_space(1)))* as1_v4i_ptr;   // global
typedef v4i __attribute__((address_space(3)))* as3_v4i_ptr;   // LDS

#define OUT_PC 64
#define IN_PC  64
#define CBIG   256            // 4 * 64 expanded channels
#define HH     64
#define WW     64
#define BB     16
#define KTOT   2304           // CBIG * 9
#define BLOCK_M 256
#define BLOCK_N 64            // one full output row (wo = 0..63)
#define KCH    32             // WMMA K per step
#define NCHUNK (KTOT / KCH)   // 72

#if defined(__AMDGCN__) && __has_builtin(__builtin_amdgcn_global_load_async_to_lds_b128)
#define HAVE_ASYNC_LDS 1
#else
#define HAVE_ASYNC_LDS 0
#endif

__device__ __forceinline__ unsigned short f2bf(float f) {
  unsigned u = __float_as_uint(f);
  u += 0x7FFFu + ((u >> 16) & 1u);   // round-to-nearest-even
  return (unsigned short)(u >> 16);
}

__device__ __forceinline__ void wait_async_lds() {
#if HAVE_ASYNC_LDS
#if __has_builtin(__builtin_amdgcn_s_wait_asynccnt)
  __builtin_amdgcn_s_wait_asynccnt(0);
#else
  asm volatile("s_wait_asynccnt 0x0" ::: "memory");
#endif
#endif
}

// --------------------------- weight prep -----------------------------------
// wgemm[oc][k] bf16, k = (kh*3+kw)*256 + (qi*64+ci); Hamilton sign/component.
__global__ void qconv_prep_weights(const float* __restrict__ wr,
                                   const float* __restrict__ wi,
                                   const float* __restrict__ wj,
                                   const float* __restrict__ wk,
                                   unsigned short* __restrict__ wgemm) {
  int idx = blockIdx.x * blockDim.x + threadIdx.x;
  if (idx >= CBIG * KTOT) return;
  int oc  = idx / KTOT;
  int k   = idx % KTOT;
  int p   = k / CBIG;          // kernel position 0..8
  int icq = k % CBIG;
  int kh = p / 3, kw = p % 3;
  int qo = oc >> 6, oi = oc & 63;
  int qi = icq >> 6, ci = icq & 63;
  // [[ r,-i,-j,-k],[ i, r,-k, j],[ j, k, r,-i],[ k,-j, i, r]]
  const int   comp[4][4] = {{0,1,2,3},{1,0,3,2},{2,3,0,1},{3,2,1,0}};
  const float sgn [4][4] = {{1.f,-1.f,-1.f,-1.f},
                            {1.f, 1.f,-1.f, 1.f},
                            {1.f, 1.f, 1.f,-1.f},
                            {1.f,-1.f, 1.f, 1.f}};
  const float* src;
  switch (comp[qo][qi]) {
    case 0:  src = wr; break;
    case 1:  src = wi; break;
    case 2:  src = wj; break;
    default: src = wk; break;
  }
  float v = sgn[qo][qi] * src[((oi * IN_PC + ci) * 3 + kh) * 3 + kw];
  wgemm[idx] = f2bf(v);
}

// --------------------------- main GEMM kernel ------------------------------
__global__ __launch_bounds__(256)
void qconv_wmma(const float* __restrict__ x,            // [B,64,H,W,4] f32
                const unsigned short* __restrict__ wgemm,
                const float* __restrict__ bias_r,       // [64]
                float* __restrict__ out) {              // [B,64,H,W,4] f32
  __shared__ unsigned short ldsA[BLOCK_M * KCH];  // 256x32 bf16 = 16 KB
  __shared__ unsigned short ldsB[BLOCK_N * KCH];  //  64x32 bf16 =  4 KB

  const int tid  = threadIdx.x;
  const int lane = tid & 31;
  const int wv   = tid >> 5;        // wave 0..7
  const int wm   = wv >> 1;         // 0..3  (M direction, 64 rows each)
  const int wn   = wv & 1;          // 0..1  (N direction, 32 cols each)

  const int rowTile = blockIdx.x;   // b*H + ho
  const int b   = rowTile >> 6;
  const int ho  = rowTile & 63;

  f32x8 acc[4][2] = {};

  // staging roles (loop-invariant): A row per thread, B pixel per thread
  const int sB_jg = tid >> 6;            // 0..3 channel-octet group
  const int sB_n  = tid & 63;            // pixel in row == wo

  for (int kc = 0; kc < NCHUNK; ++kc) {
    int p     = kc >> 3;               // kernel position for this chunk
    int cbase = (kc & 7) * 32;         // expanded-channel base
    int kh = p / 3, kw = p % 3;
    int hi = ho + kh - 1;
    int wi = sB_n + kw - 1;
    bool inb = ((unsigned)hi < (unsigned)HH) & ((unsigned)wi < (unsigned)WW);

    // ---- stage A: one 64B weight row per thread (bf16, contiguous) ----
    {
      const unsigned short* srcA = wgemm + tid * KTOT + kc * KCH;
      unsigned short* dstA = ldsA + tid * KCH;
#if HAVE_ASYNC_LDS
#pragma unroll
      for (int i = 0; i < 4; ++i)
        __builtin_amdgcn_global_load_async_to_lds_b128(
            (as1_v4i_ptr)(srcA + i * 8),
            (as3_v4i_ptr)(dstA + i * 8),
            0, 0);
#else
      const uint4* s4 = (const uint4*)srcA;
      uint4* d4 = (uint4*)dstA;
#pragma unroll
      for (int i = 0; i < 4; ++i) d4[i] = s4[i];
      if (kc + 1 < NCHUNK)
        __builtin_prefetch(srcA + KCH, 0, 3);
#endif
    }
    // ---- stage B: shifted input row, 8 channels per thread, f32->bf16 ----
    {
      unsigned short tmp[8];
#pragma unroll
      for (int e = 0; e < 8; ++e) {
        int icq = cbase + sB_jg * 8 + e;     // expanded input channel
        int q = icq >> 6, c = icq & 63;      // component / packed channel
        float v = 0.f;
        if (inb) v = x[(((b * IN_PC + c) * HH + hi) * WW + wi) * 4 + q];
        tmp[e] = f2bf(v);
      }
      *(uint4*)(ldsB + sB_n * KCH + sB_jg * 8) = *(const uint4*)tmp;
    }
    wait_async_lds();
    __syncthreads();

    // ---- load fragments per CDNA5 bf16 16x16x32 layout ----
    // lanes 0-15: K 0..7 then 16..23 ; lanes 16-31: K 8..15 then 24..31
    union Frag { uint4 q[2]; bf16x16 v; };
    const int fr   = lane & 15;
    const int kofs = (lane >> 4) * 8;
    Frag afrag[4], bfrag[2];
#pragma unroll
    for (int i = 0; i < 4; ++i) {
      int row = wm * 64 + i * 16 + fr;
      afrag[i].q[0] = *(const uint4*)(ldsA + row * KCH + kofs);
      afrag[i].q[1] = *(const uint4*)(ldsA + row * KCH + kofs + 16);
    }
#pragma unroll
    for (int j = 0; j < 2; ++j) {
      int col = wn * 32 + j * 16 + fr;
      bfrag[j].q[0] = *(const uint4*)(ldsB + col * KCH + kofs);
      bfrag[j].q[1] = *(const uint4*)(ldsB + col * KCH + kofs + 16);
    }

#pragma unroll
    for (int i = 0; i < 4; ++i)
#pragma unroll
      for (int j = 0; j < 2; ++j)
        acc[i][j] = __builtin_amdgcn_wmma_f32_16x16x32_bf16(
            false, afrag[i].v, false, bfrag[j].v,
            (short)0, acc[i][j], false, false);

    __syncthreads();
  }

  // ---- store: C/D layout — VGPR v: lanes0-15 M=v, lanes16-31 M=v+8 ----
  const int mlo = ((lane >> 4) << 3);
#pragma unroll
  for (int i = 0; i < 4; ++i) {
#pragma unroll
    for (int j = 0; j < 2; ++j) {
      int wo = wn * 32 + j * 16 + (lane & 15);
#pragma unroll
      for (int v = 0; v < 8; ++v) {
        int m   = wm * 64 + i * 16 + v + mlo;   // expanded out channel 0..255
        int q   = m >> 6;
        int opc = m & 63;
        float val = acc[i][j][v];
        if (q == 0) val += bias_r[opc];         // bias on real component only
        out[(((b * OUT_PC + opc) * HH + ho) * WW + wo) * 4 + q] = val;
      }
    }
  }
}

// --------------------------- host entry ------------------------------------
extern "C" void kernel_launch(void* const* d_in, const int* in_sizes, int n_in,
                              void* d_out, int out_size, void* d_ws, size_t ws_size,
                              hipStream_t stream) {
  const float* x      = (const float*)d_in[0];
  const float* wr     = (const float*)d_in[1];
  const float* wi     = (const float*)d_in[2];
  const float* wj     = (const float*)d_in[3];
  const float* wk     = (const float*)d_in[4];
  const float* bias_r = (const float*)d_in[5];
  float* out = (float*)d_out;

  // Hamilton-expanded bf16 weight in workspace: 256*2304*2 = 1.18 MB
  unsigned short* wgemm = (unsigned short*)d_ws;

  int total = CBIG * KTOT;
  qconv_prep_weights<<<(total + 255) / 256, 256, 0, stream>>>(wr, wi, wj, wk,
                                                              wgemm);

  dim3 grid(BB * HH);                  // 1024 row tiles, M covered in-block
  qconv_wmma<<<grid, 256, 0, stream>>>(x, wgemm, bias_r, out);
}
